// MyGRU_37572373906189
// MI455X (gfx1250) — compile-verified, matrix operand order checked by reading, not run on hardware
//
#include <hip/hip_runtime.h>
#include <hip/hip_bf16.h>
#include <stdint.h>

// ---------------------------------------------------------------------------
// GRU: S=512, B=64, I=256, H=512.
//   Phase 1 (parallel): Gx[t,b, gate*H+j] = x_t @ Wx_gate.T + folded biases
//   Phase 2 (persistent, 16 WGs): per step t:
//       r = sigmoid(Gx_r + h @ Wrh.T), z = sigmoid(Gx_z + h @ Wuh.T)
//       n = tanh(Gx_n + r * (h @ Wnhh.T + ch)),  h = (1-z)*n + z*h
//   Recurrent weights + per-step h live in LDS in WMMA-fragment order
//   (contiguous 32B per lane per k-step -> trivial ds_load_b128 pairs),
//   h exchanged step-to-step through L2 (64 KB bf16, double-buffered),
//   cross-WG sync via device-scope atomic barrier (no kernel relaunch).
// ---------------------------------------------------------------------------

typedef unsigned short u16;
typedef __attribute__((ext_vector_type(16))) __bf16 v16bf;
typedef __attribute__((ext_vector_type(8)))  float  v8f;

union BFrag { v16bf v; uint4 q[2]; };

__device__ __forceinline__ u16 f2bf(float f) {
  uint32_t u = __float_as_uint(f);
  u += 0x7FFFu + ((u >> 16) & 1u);
  return (u16)(u >> 16);
}

__device__ __forceinline__ v8f wmma_bf16(const BFrag& a, const BFrag& b, v8f c) {
  return __builtin_amdgcn_wmma_f32_16x16x32_bf16(
      false, a.v, false, b.v, (short)0, c, false, false);
}

// Problem sizes
#define S_  512
#define B_  64
#define I_  256
#define H_  512
#define H3_ 1536            // 3*H (gate-major columns: r | z | n)
#define ROW_WR 770          // I+H+2
#define ROW_NI 257          // I+1
#define ROW_NH 513          // H+1
#define NWGS 16             // persistent workgroups
#define UPW  32             // hidden units per workgroup (H / NWGS)

// LDS layout (fragment-order blocks of 1024 B = 32 lanes x 32 B):
//   weights: [gate(3)][nt(2)][ks(16)] blocks   -> 98304 B
//   h stage: [mt(4)][ks(16)] blocks            -> 65536 B   (total 160 KB)
#define LDS_W_OFF 0
#define LDS_H_OFF 98304
#define LDS_TOTAL 163840

// ---------------------------------------------------------------------------
// Kernel 0: pack/convert.  Grid-stride over all ranges.
// ---------------------------------------------------------------------------
__global__ void gru_prep(const float* __restrict__ X,
                         const float* __restrict__ Wr, const float* __restrict__ br,
                         const float* __restrict__ Wu, const float* __restrict__ bu,
                         const float* __restrict__ Wni, const float* __restrict__ bni,
                         const float* __restrict__ Wnh, const float* __restrict__ bnh,
                         u16* __restrict__ Xbf,  u16* __restrict__ Wx,
                         u16* __restrict__ Wh,   float* __restrict__ cx,
                         float* __restrict__ ch, u16* __restrict__ hbf,
                         float* __restrict__ hf, int* __restrict__ bar) {
  const size_t NX  = (size_t)S_ * B_ * I_;      // 8388608
  const size_t NWX = (size_t)3 * H_ * I_;       // 393216
  const size_t NWH = (size_t)3 * H_ * H_;       // 786432
  const size_t NCX = H3_;                       // 1536
  const size_t NCH = H_;                        // 512
  const size_t NH  = (size_t)B_ * H_;           // 32768 (zero buffer 0)
  const size_t TOT = NX + NWX + NWH + NCX + NCH + NH + 1;
  for (size_t idx = blockIdx.x * (size_t)blockDim.x + threadIdx.x;
       idx < TOT; idx += (size_t)gridDim.x * blockDim.x) {
    size_t r = idx;
    if (r < NX) { Xbf[r] = f2bf(X[r]); continue; }
    r -= NX;
    if (r < NWX) {                              // Wx[gate*H+j][k] over x columns
      int j3 = (int)(r / I_), k = (int)(r % I_);
      int gate = j3 >> 9, j = j3 & (H_ - 1);
      float v;
      if      (gate == 0) v = Wr[(size_t)j * ROW_WR + k];
      else if (gate == 1) v = Wu[(size_t)j * ROW_WR + k];
      else                v = Wni[(size_t)j * ROW_NI + k];
      Wx[r] = f2bf(v);
      continue;
    }
    r -= NWX;
    if (r < NWH) {                              // Wh[gate*H+j][k] over h columns
      int j3 = (int)(r / H_), k = (int)(r % H_);
      int gate = j3 >> 9, j = j3 & (H_ - 1);
      float v;
      if      (gate == 0) v = Wr[(size_t)j * ROW_WR + (I_ + 1) + k];
      else if (gate == 1) v = Wu[(size_t)j * ROW_WR + (I_ + 1) + k];
      else                v = Wnh[(size_t)j * ROW_NH + k];
      Wh[r] = f2bf(v);
      continue;
    }
    r -= NWH;
    if (r < NCX) {                              // folded x-side constants
      int j3 = (int)r, gate = j3 >> 9, j = j3 & (H_ - 1);
      float v;
      if      (gate == 0) v = Wr[(size_t)j * ROW_WR + I_] + Wr[(size_t)j * ROW_WR + I_ + 1 + H_] + br[j];
      else if (gate == 1) v = Wu[(size_t)j * ROW_WR + I_] + Wu[(size_t)j * ROW_WR + I_ + 1 + H_] + bu[j];
      else                v = Wni[(size_t)j * ROW_NI + I_] + bni[j];
      cx[j3] = v;
      continue;
    }
    r -= NCX;
    if (r < NCH) {                              // h-side n-gate constant (inside r*(...))
      int j = (int)r;
      ch[j] = Wnh[(size_t)j * ROW_NH + H_] + bnh[j];
      continue;
    }
    r -= NCH;
    if (r < NH) { hbf[r] = 0; hf[r] = 0.0f; continue; }   // h0 = 0 (buffer 0)
    *bar = 0;
  }
}

// ---------------------------------------------------------------------------
// Kernel 1: Gx = Xbf @ Wx.T + cx.   [32768 x 256] @ [256 x 1536] -> fp32.
// Block = 256 threads = 8 waves; tile 64(M) x 128(N); wave w: mt=w&3, ng=w>>2,
// 4 N-tiles per wave reusing one A fragment per K-step.
// ---------------------------------------------------------------------------
__global__ void __launch_bounds__(256)
gru_gx(const u16* __restrict__ Xbf, const u16* __restrict__ Wx,
       const float* __restrict__ cx, float* __restrict__ Gx) {
  const int tid  = threadIdx.x;
  const int wave = tid >> 5, lane = tid & 31;
  const int mt = wave & 3, ng = wave >> 2;
  const int M0 = blockIdx.x * 64 + mt * 16;
  const int N0 = blockIdx.y * 128 + ng * 64;
  const int lrow = lane & 15;
  const int lk8  = (lane >> 4) * 8;       // K sub-chunk base for this lane
  const int mrow = (lane >> 4) * 8;       // C-fragment M base for this lane

  v8f acc[4] = {{}, {}, {}, {}};
  const u16* arow = Xbf + (size_t)(M0 + lrow) * I_;
#pragma unroll 2
  for (int kk = 0; kk < I_; kk += 32) {
    BFrag a;
    const uint4* pa = (const uint4*)(arow + kk + lk8);
    a.q[0] = pa[0]; a.q[1] = pa[2];       // K chunks [lk8, lk8+8) and +16
#pragma unroll
    for (int i = 0; i < 4; ++i) {
      BFrag b;
      const uint4* pb = (const uint4*)(Wx + (size_t)(N0 + i * 16 + lrow) * I_ + kk + lk8);
      b.q[0] = pb[0]; b.q[1] = pb[2];
      acc[i] = wmma_bf16(a, b, acc[i]);
    }
  }
#pragma unroll
  for (int i = 0; i < 4; ++i) {
    int col = N0 + i * 16 + lrow;
    float cb = cx[col];
#pragma unroll
    for (int r = 0; r < 8; ++r) {
      int row = M0 + mrow + r;
      Gx[(size_t)row * H3_ + col] = acc[i][r] + cb;
    }
  }
}

// ---------------------------------------------------------------------------
// Kernel 2: persistent recurrence. 16 WGs x 256 threads. WG g owns hidden
// units [g*32, g*32+32). LDS holds its 3x32x512 bf16 recurrent weight slab
// and the per-step h stage, both pre-swizzled into WMMA-fragment order.
// ---------------------------------------------------------------------------
__global__ void __launch_bounds__(256)
gru_rec(const float* __restrict__ Gx, const u16* __restrict__ Wh,
        const float* __restrict__ ch, u16* __restrict__ hbf,
        float* __restrict__ hf, float* __restrict__ out,
        int* __restrict__ bar) {
  extern __shared__ char smem[];
  const int g   = blockIdx.x;             // 0..15
  const int tid = threadIdx.x;
  const int wave = tid >> 5, lane = tid & 31;

  // One-time: stage recurrent weight slab into LDS in fragment order.
  // dst uint4 index i decomposes as: c=i&1, lane_s=(i>>1)&31, ks=(i>>6)&15,
  // nt_s=(i>>10)&1, gate=i>>11 -- here we iterate in SOURCE order for
  // coalescing and compute the destination index.
  {
    const uint4* gsrc = (const uint4*)Wh;            // 64 uint4 per 512-elem row
    uint4* wdst = (uint4*)(smem + LDS_W_OFF);
    for (int i = tid; i < 96 * 64; i += 256) {
      int row = i >> 6, off = i & 63;                // row = gate*32 + u
      int gate = row >> 5, u = row & 31;
      int ks = off >> 2;
      int lane_hi = off & 1;                         // which 8-k sub-chunk
      int c = (off >> 1) & 1;                        // which 16B chunk
      int nt_s = u >> 4;
      int lane_s = lane_hi * 16 + (u & 15);
      uint4 v = gsrc[((size_t)gate * H_ + g * UPW + u) * 64 + off];
      wdst[(((gate * 2 + nt_s) * 16 + ks) * 64) + lane_s * 2 + c] = v;
    }
  }
  __syncthreads();

  const int mt = wave & 3, nt = wave >> 2;           // 4 M-tiles x 2 N-tiles
  const int mbase = mt * 16;
  const int lrow = lane & 15;
  const int mrow = (lane >> 4) * 8;
  const int j = g * UPW + nt * 16 + lrow;            // this lane's output column
  const float chv = ch[j];

  const char* abase = smem + LDS_H_OFF + (size_t)mt * 16 * 1024 + lane * 32;
  const char* bbase_r = smem + LDS_W_OFF + (size_t)((0 * 2 + nt) * 16) * 1024 + lane * 32;
  const char* bbase_z = smem + LDS_W_OFF + (size_t)((1 * 2 + nt) * 16) * 1024 + lane * 32;
  const char* bbase_n = smem + LDS_W_OFF + (size_t)((2 * 2 + nt) * 16) * 1024 + lane * 32;

  uint4* hstage = (uint4*)(smem + LDS_H_OFF);

  for (int t = 0; t < S_; ++t) {
    const int cur = t & 1, nxt = cur ^ 1;
    const u16*   hc  = hbf + (size_t)cur * (B_ * H_);
    const float* hfc = hf  + (size_t)cur * (B_ * H_);

    // ---- Stage h (64 KB) into LDS in fragment order; linear coalesced src.
    {
      const uint4* hsrc = (const uint4*)hc;          // 4096 uint4 total
#pragma unroll
      for (int it = 0; it < 16; ++it) {
        int i = tid + it * 256;
        int b = i >> 6, off = i & 63;
        int ks = off >> 2;
        int lane_hi = off & 1;
        int c = (off >> 1) & 1;
        int mt_s = b >> 4;
        int lane_s = lane_hi * 16 + (b & 15);
        hstage[((mt_s * 16 + ks) * 64) + lane_s * 2 + c] = hsrc[i];
      }
    }

    // ---- Preload this step's Gx pre-activations + h_old into registers
    //      (loads stay outstanding across the K-loop).
    const float* gxt = Gx + (size_t)t * B_ * H3_;
    float gr[8], gz[8], gn[8], hold[8];
#pragma unroll
    for (int r = 0; r < 8; ++r) {
      const int b = mbase + mrow + r;
      gr[r]   = gxt[(size_t)b * H3_ + j];
      gz[r]   = gxt[(size_t)b * H3_ + H_ + j];
      gn[r]   = gxt[(size_t)b * H3_ + 2 * H_ + j];
      hold[r] = hfc[(size_t)b * H_ + j];
    }
    // ---- Prefetch next step's Gx lines (global_prefetch_b8).
    if (t + 1 < S_) {
      const float* gxn = gxt + (size_t)B_ * H3_;
#pragma unroll
      for (int r = 0; r < 8; ++r) {
        const int b = mbase + mrow + r;
        __builtin_prefetch(gxn + (size_t)b * H3_ + j, 0, 3);
        __builtin_prefetch(gxn + (size_t)b * H3_ + H_ + j, 0, 3);
        __builtin_prefetch(gxn + (size_t)b * H3_ + 2 * H_ + j, 0, 3);
      }
    }
    __syncthreads();   // h stage visible to all waves

    // ---- K-loop entirely out of LDS: 2 ds_load_b128 per fragment.
    v8f cr = {}, cz = {}, cn = {};
#pragma unroll 4
    for (int ks = 0; ks < 16; ++ks) {
      BFrag a, b0, b1, b2;
      const uint4* pa = (const uint4*)(abase + ks * 1024);
      const uint4* p0 = (const uint4*)(bbase_r + ks * 1024);
      const uint4* p1 = (const uint4*)(bbase_z + ks * 1024);
      const uint4* p2 = (const uint4*)(bbase_n + ks * 1024);
      a.q[0]  = pa[0]; a.q[1]  = pa[1];
      b0.q[0] = p0[0]; b0.q[1] = p0[1];
      b1.q[0] = p1[0]; b1.q[1] = p1[1];
      b2.q[0] = p2[0]; b2.q[1] = p2[1];
      cr = wmma_bf16(a, b0, cr);
      cz = wmma_bf16(a, b1, cz);
      cn = wmma_bf16(a, b2, cn);
    }

    // ---- Gate nonlinearities + blend for this lane's 8 (batch, unit) cells.
#pragma unroll
    for (int r = 0; r < 8; ++r) {
      const int b = mbase + mrow + r;
      const float rr = 1.0f / (1.0f + __expf(-(cr[r] + gr[r])));
      const float zz = 1.0f / (1.0f + __expf(-(cz[r] + gz[r])));
      const float nn = tanhf(gn[r] + rr * (cn[r] + chv));
      const float hnew = (1.0f - zz) * nn + zz * hold[r];
      out[((size_t)t * B_ + b) * H_ + j] = hnew;
      if (t == S_ - 1) out[(size_t)S_ * B_ * H_ + (size_t)b * H_ + j] = hnew;
      hf [(size_t)nxt * (B_ * H_) + (size_t)b * H_ + j] = hnew;
      hbf[(size_t)nxt * (B_ * H_) + (size_t)b * H_ + j] = f2bf(hnew);
    }

    // ---- Device-scope barrier across the 16 persistent WGs.
    __syncthreads();
    if (tid == 0) {
      __hip_atomic_fetch_add(bar, 1, __ATOMIC_RELEASE, __HIP_MEMORY_SCOPE_AGENT);
      const int target = (t + 1) * NWGS;
      while (__hip_atomic_load(bar, __ATOMIC_ACQUIRE, __HIP_MEMORY_SCOPE_AGENT) < target) {
        __builtin_amdgcn_s_sleep(2);
      }
    }
    __syncthreads();
  }
}

// ---------------------------------------------------------------------------
// Host side: workspace layout + launches (graph-capture safe).
// ---------------------------------------------------------------------------
extern "C" void kernel_launch(void* const* d_in, const int* in_sizes, int n_in,
                              void* d_out, int out_size, void* d_ws, size_t ws_size,
                              hipStream_t stream) {
  (void)in_sizes; (void)n_in; (void)out_size; (void)ws_size;
  const float* X   = (const float*)d_in[0];
  const float* Wr  = (const float*)d_in[1];
  const float* br  = (const float*)d_in[2];
  const float* Wu  = (const float*)d_in[3];
  const float* bu  = (const float*)d_in[4];
  const float* Wni = (const float*)d_in[5];
  const float* bni = (const float*)d_in[6];
  const float* Wnh = (const float*)d_in[7];
  const float* bnh = (const float*)d_in[8];
  float* out = (float*)d_out;

  char* ws = (char*)d_ws;
  const size_t GX_OFF  = 0;
  const size_t GX_B    = (size_t)S_ * B_ * H3_ * 4;          // 201326592
  const size_t XBF_OFF = GX_OFF + GX_B;
  const size_t XBF_B   = (size_t)S_ * B_ * I_ * 2;           // 16777216
  const size_t WX_OFF  = XBF_OFF + XBF_B;
  const size_t WX_B    = (size_t)3 * H_ * I_ * 2;            // 786432
  const size_t WH_OFF  = WX_OFF + WX_B;
  const size_t WH_B    = (size_t)3 * H_ * H_ * 2;            // 1572864
  const size_t CX_OFF  = WH_OFF + WH_B;
  const size_t CX_B    = (size_t)H3_ * 4;
  const size_t CH_OFF  = CX_OFF + CX_B;
  const size_t CH_B    = (size_t)H_ * 4;
  const size_t HBF_OFF = CH_OFF + CH_B;
  const size_t HBF_B   = (size_t)2 * B_ * H_ * 2;
  const size_t HF_OFF  = HBF_OFF + HBF_B;
  const size_t HF_B    = (size_t)2 * B_ * H_ * 4;
  const size_t BAR_OFF = HF_OFF + HF_B;

  float* Gx  = (float*)(ws + GX_OFF);
  u16*   Xbf = (u16*)  (ws + XBF_OFF);
  u16*   Wxp = (u16*)  (ws + WX_OFF);
  u16*   Whp = (u16*)  (ws + WH_OFF);
  float* cx  = (float*)(ws + CX_OFF);
  float* ch  = (float*)(ws + CH_OFF);
  u16*   hbf = (u16*)  (ws + HBF_OFF);
  float* hf  = (float*)(ws + HF_OFF);
  int*   bar = (int*)  (ws + BAR_OFF);

  // Phase 0: pack weights / convert X / zero h and barrier.
  gru_prep<<<4096, 256, 0, stream>>>(X, Wr, br, Wu, bu, Wni, bni, Wnh, bnh,
                                     Xbf, Wxp, Whp, cx, ch, hbf, hf, bar);
  // Phase 1: all input projections as one bf16 WMMA GEMM.
  gru_gx<<<dim3((S_ * B_) / 64, H3_ / 128), 256, 0, stream>>>(Xbf, Wxp, cx, Gx);
  // Phase 2: persistent recurrence (16 WGs, LDS-resident weights + h stage).
  gru_rec<<<NWGS, 256, LDS_TOTAL, stream>>>(Gx, Whp, ch, hbf, hf, out, bar);
}
